// UltrametricCausalSelfAttention_83442624627236
// MI455X (gfx1250) — compile-verified, matrix operand order checked by reading, not run on hardware
//
#include <hip/hip_runtime.h>
#include <hip/hip_bf16.h>
#include <stddef.h>

typedef __attribute__((ext_vector_type(16))) __bf16         v16bf;
typedef __attribute__((ext_vector_type(8)))  float          v8f;
typedef __attribute__((ext_vector_type(16))) unsigned short v16us;
typedef __attribute__((ext_vector_type(4)))  unsigned int   v4u;

#define T_DIM  2048
#define C_DIM  1024
#define H_DIM  16
#define HD_DIM 64
#define KLCP   8
#define NEGF   (-1.0e9f)

static __device__ __forceinline__ unsigned short f32_to_bf16(float f) {
  unsigned u = __builtin_bit_cast(unsigned, f);
  unsigned r = 0x7FFFu + ((u >> 16) & 1u);
  return (unsigned short)((u + r) >> 16);
}

// Async 16-byte copy global -> LDS (no VGPR round trip, tracked by ASYNCcnt).
// Generic LDS pointers carry the DS byte address in their low 32 bits.
static __device__ __forceinline__ void async_copy_b128(void* lds_dst, const void* gsrc) {
  unsigned lds_addr = (unsigned)(size_t)lds_dst;
  asm volatile("global_load_async_to_lds_b128 %0, %1, off"
               :: "v"(lds_addr), "v"(gsrc) : "memory");
}
static __device__ __forceinline__ void wait_async0() {
  asm volatile("s_wait_asynccnt 0" ::: "memory");
}

// Load one 16x32 fragment for this lane: two contiguous 16B runs
// (k-run at +0, second half-K run at +16 elements). Lowers to 2x b128 loads.
static __device__ __forceinline__ v16us ld_frag(const unsigned short* p) {
  union { v4u v[2]; v16us s; } u;
  u.v[0] = *(const v4u*)p;
  u.v[1] = *(const v4u*)(p + 16);
  return u.s;
}

static __device__ __forceinline__ v8f wmma_bf16(v16us a, v16us b, v8f c) {
  return __builtin_amdgcn_wmma_f32_16x16x32_bf16(
      false, __builtin_bit_cast(v16bf, a),
      false, __builtin_bit_cast(v16bf, b),
      (short)0, c, false, false);
}

// ---------------------------------------------------------------- cvt f32->bf16
__global__ __launch_bounds__(256)
void cvt_f32_bf16_kernel(const float* __restrict__ src, unsigned short* __restrict__ dst, int n) {
  int i = blockIdx.x * 256 + threadIdx.x;
  if (i < n) dst[i] = f32_to_bf16(src[i]);
}

// ---------------------------------------------------------------- WMMA GEMM
// C[2048,1024] = A[2048,1024] * B[1024,1024]; bf16 in, f32 out.
// Block tile 128x64, BK=64, 8 waves (4x2), 32x32 per wave.
// Double-buffered LDS: async A-copy + B prefetch overlap the WMMA work.
#define BM 128
#define BN 64
#define BK 64

__global__ __launch_bounds__(256)
void gemm_bf16_wmma_kernel(const unsigned short* __restrict__ A,
                           const unsigned short* __restrict__ B,
                           float* __restrict__ C) {
  __shared__ __align__(16) unsigned short As[2][BM * BK];  // [m][k] 2x16KB
  __shared__ __align__(16) unsigned short Bt[2][BN * BK];  // [n][k] 2x8KB (transposed)
  const int lane = threadIdx.x & 31;
  const int wave = threadIdx.x >> 5;
  const int wm = wave & 3, wn = wave >> 2;
  const int nn = lane & 15, half = (lane >> 4) & 1;
  const int brow = blockIdx.x * BM, bcol = blockIdx.y * BN;

  v4u breg[2];
  auto issueA = [&](int k0, int b) {
#pragma unroll
    for (int it = 0; it < (BM * BK) / (8 * 256); ++it) {  // 4
      int u = threadIdx.x + it * 256;
      int r = u >> 3, c8 = (u & 7) << 3;
      async_copy_b128(&As[b][u * 8], &A[(size_t)(brow + r) * C_DIM + k0 + c8]);
    }
  };
  auto issueB = [&](int k0) {
#pragma unroll
    for (int it = 0; it < (BK * BN) / (8 * 256); ++it) {  // 2
      int u = threadIdx.x + it * 256;
      int r = u >> 3, n8 = (u & 7) << 3;
      breg[it] = *(const v4u*)&B[(size_t)(k0 + r) * C_DIM + bcol + n8];
    }
  };
  auto storeB = [&](int b) {  // transpose into [n][k]
#pragma unroll
    for (int it = 0; it < (BK * BN) / (8 * 256); ++it) {
      int u = threadIdx.x + it * 256;
      int r = u >> 3, n8 = (u & 7) << 3;
      union { v4u v; unsigned short s[8]; } t;
      t.v = breg[it];
#pragma unroll
      for (int e = 0; e < 8; ++e) Bt[b][(n8 + e) * BK + r] = t.s[e];
    }
  };

  v8f acc[2][2];
#pragma unroll
  for (int ti = 0; ti < 2; ++ti)
#pragma unroll
    for (int tj = 0; tj < 2; ++tj)
#pragma unroll
      for (int j = 0; j < 8; ++j) acc[ti][tj][j] = 0.0f;

  // prologue: fill buffer 0
  issueA(0, 0);
  issueB(0);
  storeB(0);
  wait_async0();
  __syncthreads();

  int buf = 0;
  for (int k0 = 0; k0 < C_DIM; k0 += BK) {
    const int nb = buf ^ 1;
    const bool more = (k0 + BK) < C_DIM;
    if (more) { issueA(k0 + BK, nb); issueB(k0 + BK); }  // overlap with compute

#pragma unroll
    for (int s = 0; s < 2; ++s) {
      v16us af[2], bfr[2];
#pragma unroll
      for (int ti = 0; ti < 2; ++ti)
        af[ti] = ld_frag(&As[buf][(wm * 32 + ti * 16 + nn) * BK + s * 32 + half * 8]);
#pragma unroll
      for (int tj = 0; tj < 2; ++tj)
        bfr[tj] = ld_frag(&Bt[buf][(wn * 32 + tj * 16 + nn) * BK + s * 32 + half * 8]);
#pragma unroll
      for (int ti = 0; ti < 2; ++ti)
#pragma unroll
        for (int tj = 0; tj < 2; ++tj)
          acc[ti][tj] = wmma_bf16(af[ti], bfr[tj], acc[ti][tj]);
    }

    if (more) storeB(nb);
    wait_async0();
    __syncthreads();
    buf = nb;
  }

  // epilogue: constant-offset stores
  float* cp = C + (size_t)(brow + wm * 32 + half * 8) * C_DIM + bcol + wn * 32 + nn;
#pragma unroll
  for (int ti = 0; ti < 2; ++ti)
#pragma unroll
    for (int tj = 0; tj < 2; ++tj)
#pragma unroll
      for (int j = 0; j < 8; ++j)
        cp[(size_t)(ti * 16 + j) * C_DIM + tj * 16] = acc[ti][tj][j];
}

// ---------------------------------------------------------------- qd / kd skinny projections
__global__ __launch_bounds__(256)
void dproj_kernel(const float* __restrict__ x, const float* __restrict__ Wdq,
                  const float* __restrict__ Wdk, float* __restrict__ qd,
                  float* __restrict__ kd) {
  int tid = blockIdx.x * 256 + threadIdx.x;  // T * 16
  int t = tid >> 4, j = tid & 15;
  const float* W = (j < 8) ? Wdq : Wdk;
  int col = j & 7;
  const float* xr = x + (size_t)t * C_DIM;
  float acc = 0.0f;
  for (int c = 0; c < C_DIM; ++c) acc += xr[c] * W[c * 8 + col];
  ((j < 8) ? qd : kd)[(size_t)t * 8 + col] = acc;
}

// ---------------------------------------------------------------- causal & LCP mask bit-words
__global__ __launch_bounds__(64)
void lcpmask_kernel(const float* __restrict__ qd, const float* __restrict__ kd,
                    unsigned* __restrict__ mw) {
  int q = blockIdx.x;   // query row
  int w = threadIdx.x;  // key word 0..63
  float qv[8];
#pragma unroll
  for (int i = 0; i < 8; ++i) qv[i] = qd[(size_t)q * 8 + i];
  unsigned bits = 0;
  for (int t = 0; t < 32; ++t) {
    int key = w * 32 + t;
    if (key <= q) {
      const float* kv = kd + (size_t)key * 8;
      int run = 1, l = 0;
#pragma unroll
      for (int i = 0; i < 8; ++i) { run &= (qv[i] == kv[i]) ? 1 : 0; l += run; }
      if (l >= KLCP - 1) bits |= (1u << t);
    }
  }
  mw[(size_t)q * (T_DIM / 32) + w] = bits;
}

// ---------------------------------------------------------------- RoPE + RMSNorm, f32 -> bf16
__global__ __launch_bounds__(256)
void rope_rms_kernel(const float* __restrict__ in, const float* __restrict__ cosb,
                     const float* __restrict__ sinb, unsigned short* __restrict__ out) {
  int lane = threadIdx.x & 31;
  int wid = blockIdx.x * 8 + (threadIdx.x >> 5);  // over T*H
  int t = wid >> 4, h = wid & 15;
  const float* r = in + (size_t)t * C_DIM + h * HD_DIM;
  float x1 = r[lane], x2 = r[lane + 32];
  float c = cosb[(size_t)t * 32 + lane], s = sinb[(size_t)t * 32 + lane];
  float o1 = x1 * c - x2 * s;
  float o2 = x1 * s + x2 * c;
  float ss = o1 * o1 + o2 * o2;
#pragma unroll
  for (int off = 1; off < 32; off <<= 1) ss += __shfl_xor(ss, off, 32);
  float inv = rsqrtf(ss / (float)HD_DIM + 1e-6f);
  unsigned short* o = out + (size_t)t * C_DIM + h * HD_DIM;
  o[lane]      = f32_to_bf16(o1 * inv);
  o[lane + 32] = f32_to_bf16(o2 * inv);
}

// ---------------------------------------------------------------- flash attention, WMMA bf16
// Block = 8 waves = 8 query tiles of one head; double-buffered K (async LDS
// copy) and V (register->transposed LDS) tiles overlap the next key block's
// fetch with the current block's WMMA/softmax work. Finite NEG + full key loop
// reproduces the reference's uniform softmax on fully-masked rows.
__global__ __launch_bounds__(256)
void attn_wmma_kernel(const unsigned short* __restrict__ qn,
                      const unsigned short* __restrict__ kn,
                      const unsigned short* __restrict__ vb,
                      const unsigned* __restrict__ mw, float* __restrict__ y) {
  __shared__ __align__(16) unsigned short Ks[2][32 * 64];    // [key][d]  2x4KB
  __shared__ __align__(16) unsigned short Vt[2][64 * 32];    // [d][key]  2x4KB
  __shared__ __align__(16) unsigned short Plds[8][16 * 32];  // per-wave  8KB
  const int lane = threadIdx.x & 31;
  const int wave = threadIdx.x >> 5;
  const int h  = blockIdx.x >> 4;  // 8 | 128 => same h per block
  const int qb = ((blockIdx.x & 15) << 3) + wave;
  const int nn = lane & 15, half = (lane >> 4) & 1;
  unsigned short* P = &Plds[wave][0];

  const int fr = threadIdx.x >> 3, fc8 = (threadIdx.x & 7) << 3;
  v4u vreg;
  auto issueK = [&](int kb, int b) {
    async_copy_b128(&Ks[b][threadIdx.x * 8],
                    &kn[(size_t)(kb + fr) * C_DIM + h * HD_DIM + fc8]);
  };
  auto loadV = [&](int kb) {
    vreg = *(const v4u*)&vb[(size_t)(kb + fr) * C_DIM + h * HD_DIM + fc8];
  };
  auto storeV = [&](int b) {  // transpose into [d][key]
    union { v4u v; unsigned short s[8]; } t;
    t.v = vreg;
#pragma unroll
    for (int e = 0; e < 8; ++e) Vt[b][(fc8 + e) * 32 + fr] = t.s[e];
  };

  // q A-fragments: two 16x32 slabs, contiguous per lane -> global b128 loads
  v16us qa[2];
  {
    const unsigned short* qrow = qn + (size_t)(qb * 16 + nn) * C_DIM + h * HD_DIM;
    qa[0] = ld_frag(qrow + half * 8);
    qa[1] = ld_frag(qrow + 32 + half * 8);
  }

  float m_run[8], l_run[8];
  v8f o[4];
#pragma unroll
  for (int j = 0; j < 8; ++j) { m_run[j] = -3.0e38f; l_run[j] = 0.0f; }
#pragma unroll
  for (int nc = 0; nc < 4; ++nc)
#pragma unroll
    for (int j = 0; j < 8; ++j) o[nc][j] = 0.0f;

  // prologue: fill buffer 0
  issueK(0, 0);
  loadV(0);
  storeV(0);
  wait_async0();
  __syncthreads();

  int buf = 0;
  for (int kb = 0; kb < T_DIM; kb += 32) {
    const int nb = buf ^ 1;
    const bool more = (kb + 32) < T_DIM;
    if (more) { issueK(kb + 32, nb); loadV(kb + 32); }  // overlap with compute

    // ---- S = Q * K^T (two 16x16 subtiles, K-dim 64) ----
    v8f sa[2];
#pragma unroll
    for (int c = 0; c < 2; ++c) {
#pragma unroll
      for (int j = 0; j < 8; ++j) sa[c][j] = 0.0f;
#pragma unroll
      for (int s = 0; s < 2; ++s) {
        v16us kfr = ld_frag(&Ks[buf][(c * 16 + nn) * 64 + s * 32 + half * 8]);
        sa[c] = wmma_bf16(qa[s], kfr, sa[c]);
      }
    }

    // ---- mask + online softmax; stage P as bf16 in per-wave LDS ----
#pragma unroll
    for (int j = 0; j < 8; ++j) {
      unsigned word =
          mw[(size_t)(qb * 16 + half * 8 + j) * (T_DIM / 32) + (kb >> 5)];
      float s0 = ((word >> nn) & 1u)        ? sa[0][j] * 0.125f : NEGF;
      float s1 = ((word >> (16 + nn)) & 1u) ? sa[1][j] * 0.125f : NEGF;
      float rm = fmaxf(s0, s1);
#pragma unroll
      for (int off = 1; off < 16; off <<= 1) rm = fmaxf(rm, __shfl_xor(rm, off, 32));
      float mo = m_run[j];
      float mn = fmaxf(mo, rm);
      float fac = __expf(mo - mn);
      float p0 = __expf(s0 - mn);
      float p1 = __expf(s1 - mn);
      float rs = p0 + p1;
#pragma unroll
      for (int off = 1; off < 16; off <<= 1) rs += __shfl_xor(rs, off, 32);
      l_run[j] = l_run[j] * fac + rs;
      m_run[j] = mn;
#pragma unroll
      for (int nc = 0; nc < 4; ++nc) o[nc][j] *= fac;
      int row = half * 8 + j;
      P[row * 32 + nn]      = f32_to_bf16(p0);
      P[row * 32 + 16 + nn] = f32_to_bf16(p1);
    }

    // ---- O += P(16x32) * V(32x64), V read from transposed LDS tile ----
    v16us pa = ld_frag(&P[nn * 32 + half * 8]);
#pragma unroll
    for (int nc = 0; nc < 4; ++nc) {
      v16us vf = ld_frag(&Vt[buf][(nc * 16 + nn) * 32 + half * 8]);
      o[nc] = wmma_bf16(pa, vf, o[nc]);
    }

    if (more) storeV(nb);
    wait_async0();
    __syncthreads();
    buf = nb;
  }

  // ---- normalize and store ----
  float* yp = y + (size_t)(qb * 16 + half * 8) * C_DIM + h * HD_DIM + nn;
#pragma unroll
  for (int j = 0; j < 8; ++j) {
    float inv = 1.0f / l_run[j];
#pragma unroll
    for (int nc = 0; nc < 4; ++nc)
      yp[(size_t)j * C_DIM + nc * 16] = o[nc][j] * inv;
  }
}

// ---------------------------------------------------------------- launch
extern "C" void kernel_launch(void* const* d_in, const int* in_sizes, int n_in,
                              void* d_out, int out_size, void* d_ws, size_t ws_size,
                              hipStream_t stream) {
  const float* x    = (const float*)d_in[0];
  const float* cosb = (const float*)d_in[1];
  const float* sinb = (const float*)d_in[2];
  const float* Wq   = (const float*)d_in[3];
  const float* Wk   = (const float*)d_in[4];
  const float* Wv   = (const float*)d_in[5];
  const float* Wp   = (const float*)d_in[6];
  const float* Wdq  = (const float*)d_in[7];
  const float* Wdk  = (const float*)d_in[8];

  char* ws = (char*)d_ws;
  size_t off = 0;
  auto take = [&](size_t bytes) -> char* {
    char* p = ws + off;
    off += (bytes + 255) & ~(size_t)255;
    return p;
  };
  const size_t TC = (size_t)T_DIM * C_DIM;
  const size_t CC = (size_t)C_DIM * C_DIM;

  unsigned short* xb  = (unsigned short*)take(TC * 2);
  unsigned short* wqb = (unsigned short*)take(CC * 2);
  unsigned short* wkb = (unsigned short*)take(CC * 2);
  unsigned short* wvb = (unsigned short*)take(CC * 2);
  unsigned short* wpb = (unsigned short*)take(CC * 2);
  float*          qf  = (float*)take(TC * 4);
  float*          kf  = (float*)take(TC * 4);
  float*          vf  = (float*)take(TC * 4);
  unsigned short* qnb = (unsigned short*)take(TC * 2);
  unsigned short* knb = (unsigned short*)take(TC * 2);
  unsigned short* vbb = (unsigned short*)take(TC * 2);
  float*          qdb = (float*)take((size_t)T_DIM * 8 * 4);
  float*          kdb = (float*)take((size_t)T_DIM * 8 * 4);
  unsigned*       mwb = (unsigned*)take((size_t)T_DIM * (T_DIM / 32) * 4);
  float*          yf  = (float*)take(TC * 4);
  unsigned short* ybb = (unsigned short*)take(TC * 2);

  // f32 -> bf16 conversions
  cvt_f32_bf16_kernel<<<(int)((TC + 255) / 256), 256, 0, stream>>>(x, xb, (int)TC);
  cvt_f32_bf16_kernel<<<(int)((CC + 255) / 256), 256, 0, stream>>>(Wq, wqb, (int)CC);
  cvt_f32_bf16_kernel<<<(int)((CC + 255) / 256), 256, 0, stream>>>(Wk, wkb, (int)CC);
  cvt_f32_bf16_kernel<<<(int)((CC + 255) / 256), 256, 0, stream>>>(Wv, wvb, (int)CC);
  cvt_f32_bf16_kernel<<<(int)((CC + 255) / 256), 256, 0, stream>>>(Wp, wpb, (int)CC);

  // q/k/v GEMMs (WMMA)
  dim3 gg(T_DIM / BM, C_DIM / BN);
  gemm_bf16_wmma_kernel<<<gg, 256, 0, stream>>>(xb, wqb, qf);
  gemm_bf16_wmma_kernel<<<gg, 256, 0, stream>>>(xb, wkb, kf);
  gemm_bf16_wmma_kernel<<<gg, 256, 0, stream>>>(xb, wvb, vf);

  // ultrametric mask
  dproj_kernel<<<(T_DIM * 16) / 256, 256, 0, stream>>>(x, Wdq, Wdk, qdb, kdb);
  lcpmask_kernel<<<T_DIM, 64, 0, stream>>>(qdb, kdb, mwb);

  // RoPE + RMS (q,k) and bf16 convert (v)
  rope_rms_kernel<<<(T_DIM * H_DIM) / 8, 256, 0, stream>>>(qf, cosb, sinb, qnb);
  rope_rms_kernel<<<(T_DIM * H_DIM) / 8, 256, 0, stream>>>(kf, cosb, sinb, knb);
  cvt_f32_bf16_kernel<<<(int)((TC + 255) / 256), 256, 0, stream>>>(vf, vbb, (int)TC);

  // attention (WMMA flash)
  attn_wmma_kernel<<<(H_DIM * (T_DIM / 16)) / 8, 256, 0, stream>>>(qnb, knb, vbb, mwb, yf);

  // output projection (WMMA)
  cvt_f32_bf16_kernel<<<(int)((TC + 255) / 256), 256, 0, stream>>>(yf, ybb, (int)TC);
  gemm_bf16_wmma_kernel<<<gg, 256, 0, stream>>>(ybb, wpb, (float*)d_out);
}
